// MultiHeadAttention_56272661512733
// MI455X (gfx1250) — matrix operand order repaired and checked by reading, not compiled, not run on hardware
//
#include <hip/hip_runtime.h>

typedef _Float16 half_t;
typedef __attribute__((ext_vector_type(16))) _Float16 v16h;
typedef __attribute__((ext_vector_type(8)))  _Float16 v8h;
typedef __attribute__((ext_vector_type(4)))  _Float16 v4h;
typedef __attribute__((ext_vector_type(8)))  float    v8f;
typedef __attribute__((ext_vector_type(4)))  float    v4f;

template <bool C, typename T, typename F> struct cond_t { using type = T; };
template <typename T, typename F> struct cond_t<false, T, F> { using type = F; };

// ---------------------------------------------------------------------------
// CDNA5 wave32 WMMA fragment loaders (contiguous k order: k = 16*(lane/16) + element)
// ---------------------------------------------------------------------------

// A matrix 16x32 (MxK), row-major source with leading dim lda (halves).
// lanes 0-15: M=lane, K in {0..15}
// lanes 16-31: M=lane-16, K in {16..31}
static __device__ __forceinline__ v16h frag_a_ld(const half_t* p, int lda) {
  const int lane = threadIdx.x & 31;
  const int r  = lane & 15;
  const int kh = (lane >> 4) << 4;  // 0 or 16
  v16h a;
#pragma unroll
  for (int i = 0; i < 8; ++i) {
    const int kb = 2 * i + kh;
    a[2 * i]     = p[r * lda + kb];
    a[2 * i + 1] = p[r * lda + kb + 1];
  }
  return a;
}

// B matrix 32x16 (KxN), row-major source (element (k,n) at p[k*ldb+n]).
static __device__ __forceinline__ v16h frag_b_ld(const half_t* p, int ldb) {
  const int lane = threadIdx.x & 31;
  const int n  = lane & 15;
  const int kh = (lane >> 4) << 4;  // 0 or 16
  v16h b;
#pragma unroll
  for (int i = 0; i < 8; ++i) {
    b[2 * i]     = p[(kh + 2 * i) * ldb + n];
    b[2 * i + 1] = p[(kh + 2 * i + 1) * ldb + n];
  }
  return b;
}

static __device__ __forceinline__ v8f wmma_f16(v16h a, v16h b, v8f c) {
  return __builtin_amdgcn_wmma_f32_16x16x32_f16(false, a, false, b, (short)0, c,
                                                false, false);
}

static __device__ __forceinline__ v4h cvt4(v4f f) {
  v4h h;
  h[0] = (half_t)f[0]; h[1] = (half_t)f[1];
  h[2] = (half_t)f[2]; h[3] = (half_t)f[3];
  return h;
}

static __device__ __forceinline__ v16h cat16(v8h lo, v8h hi) {
  return __builtin_shufflevector(lo, hi, 0, 1, 2, 3, 4, 5, 6, 7, 8, 9, 10, 11,
                                 12, 13, 14, 15);
}

// ---------------------------------------------------------------------------
// Generic tiled GEMM: C = A * B (+ bias).  AT/BT converted to f16 into LDS,
// f32 WMMA accumulation, CT output.  Batched via blockIdx.z with (b,h) strides.
// 256 threads = 8 waves; waves tiled (BM/WM) x (BN/WN); per-wave WMxWN tile.
// Double-buffered LDS with register staging: global loads for slab k+1 are in
// flight while WMMAs consume slab k; one barrier per K-slab.
// ---------------------------------------------------------------------------
template <typename AT, typename BT, typename CT, int BM, int BN, int WM, int WN>
__global__ __launch_bounds__(256) void wmma_gemm(
    const AT* __restrict__ A, int lda, const BT* __restrict__ Bm, int ldb,
    CT* __restrict__ C, int ldc, int K, const float* __restrict__ bias, int HB,
    long aSB, long aSH, long bSB, long bSH, long cSB, long cSH) {
  constexpr int BK = 32;
  constexpr int WAVES_N = BN / WN;
  constexpr int MI = WM / 16;
  constexpr int NI = WN / 16;
  constexpr int SA = BK + 8;  // 40 halves = 80 B (16B multiple)
  constexpr int SB = BN + 8;  // 136/72 halves = 272/144 B (16B multiples)

  __shared__ half_t sa[2][BM][SA];
  __shared__ half_t sb[2][BK][SB];

  const int z  = blockIdx.z;
  const int bb = z / HB;
  const int hh = z % HB;
  A  += (long)bb * aSB + (long)hh * aSH;
  Bm += (long)bb * bSB + (long)hh * bSH;
  C  += (long)bb * cSB + (long)hh * cSH;

  const int tid  = threadIdx.x;
  const int w    = tid >> 5;
  const int wrow = w / WAVES_N;
  const int wcol = w % WAVES_N;
  const int rowBase = blockIdx.y * BM;
  const int colBase = blockIdx.x * BN;

  // ---- hoisted cooperative-load addressing ----
  constexpr int AV  = (sizeof(AT) == 4) ? 4 : 8;   // elements per vector
  constexpr int ACH = (BM * BK) / (256 * AV);      // vectors per thread
  constexpr int AVR = BK / AV;                     // vectors per tile row
  using AReg = typename cond_t<sizeof(AT) == 4, v4f, v8h>::type;
  const AT* aPtr[ACH];
  half_t*   aDst[ACH];
  AReg      aReg[ACH];
#pragma unroll
  for (int i = 0; i < ACH; ++i) {
    const int li = tid + i * 256;
    const int r = li / AVR;
    const int c = (li % AVR) * AV;
    aPtr[i] = A + (long)(rowBase + r) * lda + c;
    aDst[i] = &sa[0][r][c];
  }
  constexpr int BV  = (sizeof(BT) == 4) ? 4 : 8;
  constexpr int BCH = (BK * BN) / (256 * BV);
  constexpr int BVR = BN / BV;
  using BReg = typename cond_t<sizeof(BT) == 4, v4f, v8h>::type;
  const BT* bPtr[BCH];
  half_t*   bDst[BCH];
  BReg      bReg[BCH];
#pragma unroll
  for (int i = 0; i < BCH; ++i) {
    const int li = tid + i * 256;
    const int r = li / BVR;
    const int c = (li % BVR) * BV;
    bPtr[i] = Bm + (long)r * ldb + colBase + c;
    bDst[i] = &sb[0][r][c];
  }
  const long bStep = (long)BK * ldb;

  v8f acc[MI][NI] = {};

  // prologue: slab 0 into registers
#pragma unroll
  for (int i = 0; i < ACH; ++i) {
    aReg[i] = *(const AReg*)aPtr[i];
    aPtr[i] += BK;
  }
#pragma unroll
  for (int i = 0; i < BCH; ++i) {
    bReg[i] = *(const BReg*)bPtr[i];
    bPtr[i] += bStep;
  }

  const int NK = K / BK;
  int buf = 0;
  for (int k = 0; k < NK; ++k) {
    // commit staged registers to LDS[buf]
#pragma unroll
    for (int i = 0; i < ACH; ++i) {
      half_t* d = aDst[i] + buf * (BM * SA);
      if constexpr (sizeof(AT) == 4) *(v4h*)d = cvt4(aReg[i]);
      else                           *(v8h*)d = aReg[i];
    }
#pragma unroll
    for (int i = 0; i < BCH; ++i) {
      half_t* d = bDst[i] + buf * (BK * SB);
      if constexpr (sizeof(BT) == 4) *(v4h*)d = cvt4(bReg[i]);
      else                           *(v8h*)d = bReg[i];
    }
    __syncthreads();

    // prefetch slab k+1 (in flight during WMMA)
    if (k + 1 < NK) {
#pragma unroll
      for (int i = 0; i < ACH; ++i) {
        aReg[i] = *(const AReg*)aPtr[i];
        aPtr[i] += BK;
      }
#pragma unroll
      for (int i = 0; i < BCH; ++i) {
        bReg[i] = *(const BReg*)bPtr[i];
        bPtr[i] += bStep;
      }
    }

    // compute slab k out of LDS[buf]
    v16h af[MI];
    v16h bf[NI];
#pragma unroll
    for (int mi = 0; mi < MI; ++mi)
      af[mi] = frag_a_ld(&sa[buf][wrow * WM + mi * 16][0], SA);
#pragma unroll
    for (int ni = 0; ni < NI; ++ni)
      bf[ni] = frag_b_ld(&sb[buf][0][wcol * WN + ni * 16], SB);
#pragma unroll
    for (int mi = 0; mi < MI; ++mi)
#pragma unroll
      for (int ni = 0; ni < NI; ++ni)
        acc[mi][ni] = wmma_f16(af[mi], bf[ni], acc[mi][ni]);

    buf ^= 1;
  }

  // Epilogue: C/D layout — VGPR r holds row (lane>>4)*8 + r, col = lane&15
  const int lane = tid & 31;
  const int n  = lane & 15;
  const int m0 = (lane >> 4) * 8;
#pragma unroll
  for (int mi = 0; mi < MI; ++mi)
#pragma unroll
    for (int ni = 0; ni < NI; ++ni) {
      const int col = colBase + wcol * WN + ni * 16 + n;
      const float bv = bias ? bias[col] : 0.0f;
      CT* cp = C + (long)(rowBase + wrow * WM + mi * 16 + m0) * ldc + col;
#pragma unroll
      for (int r = 0; r < 8; ++r) {
        cp[0] = (CT)(acc[mi][ni][r] + bv);
        cp += ldc;
      }
    }
}

// ---------------------------------------------------------------------------
// Fused scores = Q K^T * inv_scale, causal mask, row softmax -> attn (f32).
// Block = 256 threads (8 waves) owns 16 query rows of one (b,h).
// Wave w owns key columns [w*256, w*256+256): 16 accumulator fragments.
// K^T fragments are per-lane contiguous 32B runs -> loaded straight from
// global (b128 x2 per fragment); no LDS staging, no barriers in the hot loop.
// ---------------------------------------------------------------------------
__global__ __launch_bounds__(256) void attn_softmax(
    const half_t* __restrict__ Qh, const half_t* __restrict__ Kh,
    float* __restrict__ attn, int S, int D, int H, float inv_scale) {
  __shared__ half_t sq[16 * 64];
  __shared__ float redmax[8][16];
  __shared__ float redsum[8][16];

  const int tid  = threadIdx.x;
  const int lane = tid & 31;
  const int w    = tid >> 5;
  const int qbase = blockIdx.x * 16;
  const int z = blockIdx.y;
  const int b = z / H;
  const int h = z % H;

  const half_t* Qp = Qh + (long)b * S * D + h * 64;
  const half_t* Kp = Kh + (long)b * S * D + h * 64;
  float* attnZ = attn + (long)z * S * S;

  // Q strip: 16x64 halves = 128 x v8h (128-bit copies)
  if (tid < 128) {
    const int r = tid >> 3, c = (tid & 7) * 8;
    *(v8h*)&sq[r * 64 + c] = *(const v8h*)&Qp[(long)(qbase + r) * D + c];
  }
  __syncthreads();

  const v16h a0 = frag_a_ld(sq, 64);       // K-dim 0..31
  const v16h a1 = frag_a_ld(sq + 32, 64);  // K-dim 32..63

  const int n  = lane & 15;
  const int m0 = (lane >> 4) * 8;
  const int kh2 = (lane >> 4) << 4;  // 0 or 16: this lane's K-chunk in B^T

  // per-lane fragment source: key row (w*256 + t*16 + n), head-dims kh2..kh2+15
  const half_t* kl = Kp + (long)(w * 256 + n) * D + kh2;
  const int tStep = 16 * D;  // halves per 16-key tile

  v8f acc[16] = {};
#pragma unroll
  for (int t = 0; t < 16; ++t) {
    const half_t* p = kl + t * tStep;
    const v16h b0 = cat16(*(const v8h*)p,        *(const v8h*)(p + 8));
    const v16h b1 = cat16(*(const v8h*)(p + 32), *(const v8h*)(p + 40));
    v8f c = acc[t];
    c = wmma_f16(a0, b0, c);
    c = wmma_f16(a1, b1, c);
    acc[t] = c;
  }

  // scale + causal mask + per-lane row max
  float mx[8];
#pragma unroll
  for (int r = 0; r < 8; ++r) mx[r] = -__builtin_inff();
#pragma unroll
  for (int t = 0; t < 16; ++t) {
    const int j = (w * 16 + t) * 16 + n;
#pragma unroll
    for (int r = 0; r < 8; ++r) {
      const int q = qbase + m0 + r;
      float s = acc[t][r] * inv_scale;
      if (j > q) s = -__builtin_inff();
      acc[t][r] = s;
      mx[r] = fmaxf(mx[r], s);
    }
  }
  // reduce max over the 16 lanes sharing each row, then across waves via LDS
#pragma unroll
  for (int off = 1; off < 16; off <<= 1)
#pragma unroll
    for (int r = 0; r < 8; ++r) mx[r] = fmaxf(mx[r], __shfl_xor(mx[r], off, 32));
  if (n == 0) {
#pragma unroll
    for (int r = 0; r < 8; ++r) redmax[w][m0 + r] = mx[r];
  }
  __syncthreads();
  float rmax[8];
#pragma unroll
  for (int r = 0; r < 8; ++r) {
    float m = redmax[0][m0 + r];
#pragma unroll
    for (int ww = 1; ww < 8; ++ww) m = fmaxf(m, redmax[ww][m0 + r]);
    rmax[r] = m;
  }
  // exp + per-lane row sum (masked entries: exp(-inf)=0)
  float sm[8];
#pragma unroll
  for (int r = 0; r < 8; ++r) sm[r] = 0.0f;
#pragma unroll
  for (int t = 0; t < 16; ++t)
#pragma unroll
    for (int r = 0; r < 8; ++r) {
      const float p = __expf(acc[t][r] - rmax[r]);
      acc[t][r] = p;
      sm[r] += p;
    }
#pragma unroll
  for (int off = 1; off < 16; off <<= 1)
#pragma unroll
    for (int r = 0; r < 8; ++r) sm[r] += __shfl_xor(sm[r], off, 32);
  if (n == 0) {
#pragma unroll
    for (int r = 0; r < 8; ++r) redsum[w][m0 + r] = sm[r];
  }
  __syncthreads();
  float rinv[8];
#pragma unroll
  for (int r = 0; r < 8; ++r) {
    float s = redsum[0][m0 + r];
#pragma unroll
    for (int ww = 1; ww < 8; ++ww) s += redsum[ww][m0 + r];
    rinv[r] = 1.0f / s;
  }
  // normalized attention out (single write of the 512 MB tensor)
  float* rowPtr[8];
#pragma unroll
  for (int r = 0; r < 8; ++r)
    rowPtr[r] = attnZ + (long)(qbase + m0 + r) * S + w * 256 + n;
#pragma unroll
  for (int t = 0; t < 16; ++t)
#pragma unroll
    for (int r = 0; r < 8; ++r) rowPtr[r][t * 16] = acc[t][r] * rinv[r];
}

// ---------------------------------------------------------------------------
extern "C" void kernel_launch(void* const* d_in, const int* in_sizes, int n_in,
                              void* d_out, int out_size, void* d_ws,
                              size_t ws_size, hipStream_t stream) {
  (void)in_sizes; (void)n_in; (void)out_size; (void)ws_size;
  const float* query = (const float*)d_in[0];
  const float* key_  = (const float*)d_in[1];
  const float* value = (const float*)d_in[2];
  // d_in[3] = mask (tril) — causal mask applied analytically in attn_softmax
  const float* Wq = (const float*)d_in[4];
  const float* Wk = (const float*)d_in[5];
  const float* Wv = (const float*)d_in[6];
  const float* Wo = (const float*)d_in[7];
  const float* bo = (const float*)d_in[8];

  constexpr int  Bz = 2, S = 2048, D = 1024, H = 16, HD = 64;
  constexpr long MS = (long)Bz * S;  // 4096 rows

  float* out  = (float*)d_out;       // (B,S,D)
  float* attn = out + MS * D;        // (B,H,S,S)

  half_t* Qh = (half_t*)d_ws;        // f16 intermediates: 4 x 8 MB = 32 MB
  half_t* Kh = Qh + MS * D;
  half_t* Vh = Kh + MS * D;
  half_t* Xh = Vh + MS * D;

  dim3 blk(256, 1, 1);

  // Q/K/V projections: (4096x1024) @ (1024x1024) -> f16
  dim3 gp(D / 128, MS / 128, 1);
  wmma_gemm<float, float, half_t, 128, 128, 64, 32><<<gp, blk, 0, stream>>>(
      query, D, Wq, D, Qh, D, D, nullptr, 1, 0, 0, 0, 0, 0, 0);
  wmma_gemm<float, float, half_t, 128, 128, 64, 32><<<gp, blk, 0, stream>>>(
      key_, D, Wk, D, Kh, D, D, nullptr, 1, 0, 0, 0, 0, 0, 0);
  wmma_gemm<float, float, half_t, 128, 128, 64, 32><<<gp, blk, 0, stream>>>(
      value, D, Wv, D, Vh, D, D, nullptr, 1, 0, 0, 0, 0, 0, 0);

  // fused scores + causal mask + softmax -> attn (f32, written once)
  dim3 ga(S / 16, Bz * H, 1);
  attn_softmax<<<ga, blk, 0, stream>>>(Qh, Kh, attn, S, D, H, 0.125f);

  // x = attn @ V_h per (b,h): (2048x2048) @ (2048x64) -> f16, head-concat layout
  dim3 gv(1, S / 128, Bz * H);
  wmma_gemm<float, half_t, half_t, 128, 64, 32, 32><<<gv, blk, 0, stream>>>(
      attn, S, Vh, D, Xh, D, S, nullptr, H, (long)H * S * S, (long)S * S,
      (long)S * D, (long)HD, (long)S * D, (long)HD);

  // out = x @ Wo + bo
  wmma_gemm<half_t, float, float, 128, 128, 64, 32><<<gp, blk, 0, stream>>>(
      Xh, D, Wo, D, out, D, D, bo, 1, 0, 0, 0, 0, 0, 0);
}
